// DecoderRNN_2791728742402
// MI455X (gfx1250) — compile-verified, matrix operand order checked by reading, not run on hardware
//
#include <hip/hip_runtime.h>
#include <hip/hip_bf16.h>
#include <stdint.h>

// Problem constants (reference): V=32000, H=512, L=1, B=16, S=64
#define VV   32000
#define HH   512
#define BB   16
#define SS   64
#define H2   1024   // 2H
#define H3   1536   // 3H
#define MROWS 1024  // B*S

typedef __attribute__((ext_vector_type(16))) __bf16 v16bf;
typedef __attribute__((ext_vector_type(8)))  __bf16 v8bf;
typedef __attribute__((ext_vector_type(8)))  float  v8f;
typedef __attribute__((ext_vector_type(4)))  float  f32x4;

// ---------------------------------------------------------------------------
// Fragment loaders (ISA 7.12.2 16-bit A/B layout: lanes 0-15 hold K=[k0..k0+7]
// and K=[k0+16..k0+23]; lanes 16-31 hold K=[k0+8..k0+15] and K=[k0+24..k0+31])
// ---------------------------------------------------------------------------
__device__ __forceinline__ v16bf load_frag_bf16(const __bf16* rowptr, int k0, int lane) {
    const int kk = k0 + ((lane & 16) ? 8 : 0);
    const v8bf lo = *(const v8bf*)(rowptr + kk);
    const v8bf hi = *(const v8bf*)(rowptr + kk + 16);
    v16bf f;
#pragma unroll
    for (int i = 0; i < 8; ++i) { f[i] = lo[i]; f[i + 8] = hi[i]; }
    return f;
}

// Load fp32 row-major weights, convert to bf16 in registers (v_cvt_pk_bf16_f32)
__device__ __forceinline__ v16bf load_frag_f32(const float* rowptr, int k0, int lane) {
    const int kk = k0 + ((lane & 16) ? 8 : 0);
    const f32x4 a0 = *(const f32x4*)(rowptr + kk + 0);
    const f32x4 a1 = *(const f32x4*)(rowptr + kk + 4);
    const f32x4 b0 = *(const f32x4*)(rowptr + kk + 16);
    const f32x4 b1 = *(const f32x4*)(rowptr + kk + 20);
    v16bf f;
#pragma unroll
    for (int i = 0; i < 4; ++i) {
        f[i]      = (__bf16)a0[i];
        f[i + 4]  = (__bf16)a1[i];
        f[i + 8]  = (__bf16)b0[i];
        f[i + 12] = (__bf16)b1[i];
    }
    return f;
}

// ---------------------------------------------------------------------------
// Kernel 1: gather embedding rows + broadcast thought, convert to bf16.
// X row r = s*B + b (matches gi's [S,B,*] time-major ordering), 2H columns.
// ---------------------------------------------------------------------------
__global__ void prep_x_kernel(const int* __restrict__ tokens,
                              const float* __restrict__ thought,
                              const float* __restrict__ emb,
                              __bf16* __restrict__ X) {
    const int r = blockIdx.x;          // 0..1023
    const int s = r >> 4;              // r / B
    const int b = r & 15;              // r % B
    const int tok = tokens[b * SS + s];            // target_seqs [B,S]
    const float* e  = emb + (size_t)tok * HH;
    const float* th = thought + (size_t)b * HH;    // thought [1,B,H]
    __bf16* xr = X + (size_t)r * H2;
    for (int c = threadIdx.x; c < HH; c += blockDim.x) {
        xr[c]      = (__bf16)e[c];
        xr[HH + c] = (__bf16)th[c];
    }
}

// ---------------------------------------------------------------------------
// WMMA GEMM: C[M x N] = A[M x K](bf16,row-major) * W[N x K](f32,row-major)^T + bias
// Block = 256 threads = 8 waves arranged 2(M) x 4(N); each wave: 1 M-tile x NT N-tiles.
// The block's A tile (32 x K bf16) is staged into LDS with ASYNC global->LDS
// copies (ASYNCcnt path) so WMMA A-fragments come from ds_load_b128 while the
// much larger W stream flows VMEM->VGPR with in-register bf16 conversion.
// grid.x = M/32, grid.y = N/(64*NT); dynamic LDS = 32*K*2 bytes.
// ---------------------------------------------------------------------------
template <int KDIM, int NT>
__global__ void gemm_bf16_kernel(const __bf16* __restrict__ A,
                                 const float* __restrict__ W,
                                 const float* __restrict__ bias,
                                 float* __restrict__ C, int Ncols) {
    extern __shared__ char gsmem[];
    __bf16* Ash = (__bf16*)gsmem;                // [32][KDIM]

    const int lane = threadIdx.x & 31;
    const int wave = threadIdx.x >> 5;   // 0..7
    const int wm   = wave >> 2;          // 0..1
    const int wn   = wave & 3;           // 0..3
    const int m0   = (blockIdx.x * 2 + wm) * 16;
    const int n0   = (blockIdx.y * 4 + wn) * NT * 16;
    const int l15  = lane & 15;

    // --- async-stage A tile: 32 rows x KDIM bf16, 16B chunks ---
    {
        const int chunks = (32 * KDIM * 2) / 16;               // 16B units
        const uint32_t lds0 = (uint32_t)(uintptr_t)Ash;        // LDS byte offset
        const char* gA = (const char*)(A + (size_t)(blockIdx.x * 32) * KDIM);
        for (int c = threadIdx.x; c < chunks; c += 256) {
            const uint32_t loff = lds0 + (uint32_t)c * 16u;
            const char*    gp   = gA + (size_t)c * 16u;
            asm volatile("global_load_async_to_lds_b128 %0, %1, off"
                         :: "v"(loff), "v"(gp) : "memory");
        }
        asm volatile("s_wait_asynccnt 0x0" ::: "memory");
        __syncthreads();
    }

    v8f acc[NT];
#pragma unroll
    for (int t = 0; t < NT; ++t) acc[t] = (v8f){0.f,0.f,0.f,0.f,0.f,0.f,0.f,0.f};

    const __bf16* Arow = Ash + (size_t)(wm * 16 + l15) * KDIM;   // LDS
    const float*  Wrow[NT];
#pragma unroll
    for (int t = 0; t < NT; ++t)
        Wrow[t] = W + (size_t)(n0 + t * 16 + l15) * KDIM;

    for (int k = 0; k < KDIM; k += 32) {
        const v16bf a = load_frag_bf16(Arow, k, lane);           // ds_load_b128
#pragma unroll
        for (int t = 0; t < NT; ++t) {
            __builtin_prefetch((const void*)(Wrow[t] + k + 32), 0, 1);
            const v16bf b = load_frag_f32(Wrow[t], k, lane);
            acc[t] = __builtin_amdgcn_wmma_f32_16x16x32_bf16(
                false, a, false, b, (short)0, acc[t], false, false);
        }
    }

    // C/D layout: element e -> m = e + 8*(lane>=16), n = lane&15
    const int mb = (lane & 16) ? 8 : 0;
#pragma unroll
    for (int t = 0; t < NT; ++t) {
        const int ncol = n0 + t * 16 + l15;
        const float bv = bias[ncol];
#pragma unroll
        for (int e = 0; e < 8; ++e)
            C[(size_t)(m0 + mb + e) * Ncols + ncol] = acc[t][e] + bv;
    }
}

// ---------------------------------------------------------------------------
// Kernel 3: GRU recurrence. One 1024-thread block (32 waves). Hidden state
// lives in LDS (fp32 + bf16 copies); per step gh = h * W_hh^T via WMMA
// (M=16, N=1536 -> 96 tiles, 3 per wave; K=512). gh goes to dynamic LDS.
// ---------------------------------------------------------------------------
__global__ void gru_kernel(const float* __restrict__ gi,     // [S*B, 3H], b_ih folded in
                           const float* __restrict__ W_hh,   // [3H, H]
                           const float* __restrict__ b_hh,   // [3H]
                           const float* __restrict__ h0,     // hidden[0]: [B, H]
                           __bf16* __restrict__ ys,          // [B*S, H] bf16 (row = b*S+s)
                           float* __restrict__ hlast) {      // [B, H]
    __shared__ float  h_s[BB * HH];     // 32 KB
    __shared__ __bf16 hb_s[BB * HH];    // 16 KB
    extern __shared__ char dynsmem[];   // 96 KB
    float* gh_s = (float*)dynsmem;      // [16, 1536]

    const int tid  = threadIdx.x;
    const int lane = tid & 31;
    const int wave = tid >> 5;          // 0..31
    const int l15  = lane & 15;
    const int mb   = (lane & 16) ? 8 : 0;

    for (int i = tid; i < BB * HH; i += blockDim.x) {
        const float v = h0[i];
        h_s[i]  = v;
        hb_s[i] = (__bf16)v;
    }
    __syncthreads();

    for (int s = 0; s < SS; ++s) {
        // --- gh = h * W_hh^T : wave handles N-tiles [wave*3 .. wave*3+2] ---
        v8f acc[3];
#pragma unroll
        for (int t = 0; t < 3; ++t) acc[t] = (v8f){0.f,0.f,0.f,0.f,0.f,0.f,0.f,0.f};
        const __bf16* Arow = hb_s + (size_t)l15 * HH;
        const float* Wrow[3];
#pragma unroll
        for (int t = 0; t < 3; ++t)
            Wrow[t] = W_hh + (size_t)((wave * 3 + t) * 16 + l15) * HH;

        for (int k = 0; k < HH; k += 32) {
            const v16bf a = load_frag_bf16(Arow, k, lane);   // ds_load from LDS
#pragma unroll
            for (int t = 0; t < 3; ++t) {
                const v16bf b = load_frag_f32(Wrow[t], k, lane);
                acc[t] = __builtin_amdgcn_wmma_f32_16x16x32_bf16(
                    false, a, false, b, (short)0, acc[t], false, false);
            }
        }
#pragma unroll
        for (int t = 0; t < 3; ++t) {
            const int ncol = (wave * 3 + t) * 16 + l15;
#pragma unroll
            for (int e = 0; e < 8; ++e)
                gh_s[(mb + e) * H3 + ncol] = acc[t][e];
        }
        __syncthreads();

        // --- gates: 8192 elements, 8 per thread ---
        for (int i = tid; i < BB * HH; i += blockDim.x) {
            const int b = i >> 9;       // /H
            const int j = i & (HH - 1);
            const float* git = gi + (size_t)(s * BB + b) * H3;
            const float i_r = git[j];
            const float i_z = git[HH + j];
            const float i_n = git[2 * HH + j];
            const float h_r = gh_s[b * H3 + j]          + b_hh[j];
            const float h_z = gh_s[b * H3 + HH + j]     + b_hh[HH + j];
            const float h_n = gh_s[b * H3 + 2 * HH + j] + b_hh[2 * HH + j];
            const float r = 1.f / (1.f + __expf(-(i_r + h_r)));
            const float z = 1.f / (1.f + __expf(-(i_z + h_z)));
            const float n = tanhf(i_n + r * h_n);
            const float hn = (1.f - z) * n + z * h_s[i];
            h_s[i]  = hn;
            hb_s[i] = (__bf16)hn;
            ys[(size_t)(b * SS + s) * HH + j] = (__bf16)hn;  // [B,S,H] row-major
        }
        __syncthreads();
    }

    for (int i = tid; i < BB * HH; i += blockDim.x) hlast[i] = h_s[i];
}

// ---------------------------------------------------------------------------
// Kernel 5: per-row max + log-sum-exp over V (one block per row)
// ---------------------------------------------------------------------------
__global__ void row_lse_kernel(const float* __restrict__ logits,
                               float* __restrict__ rmax, float* __restrict__ rlse) {
    const int r = blockIdx.x;
    const float* p = logits + (size_t)r * VV;
    const int lane = threadIdx.x & 31, wv = threadIdx.x >> 5;
    __shared__ float sred[8];

    float m = -3.4e38f;
    for (int i = threadIdx.x; i < VV; i += 256) m = fmaxf(m, p[i]);
#pragma unroll
    for (int off = 16; off > 0; off >>= 1) m = fmaxf(m, __shfl_xor(m, off, 32));
    if (lane == 0) sred[wv] = m;
    __syncthreads();
    m = sred[0];
#pragma unroll
    for (int i = 1; i < 8; ++i) m = fmaxf(m, sred[i]);
    __syncthreads();

    float sum = 0.f;
    for (int i = threadIdx.x; i < VV; i += 256) sum += __expf(p[i] - m);
#pragma unroll
    for (int off = 16; off > 0; off >>= 1) sum += __shfl_xor(sum, off, 32);
    if (lane == 0) sred[wv] = sum;
    __syncthreads();
    if (threadIdx.x == 0) {
        float tot = 0.f;
#pragma unroll
        for (int i = 0; i < 8; ++i) tot += sred[i];
        rmax[r] = m;
        rlse[r] = logf(tot);
    }
}

// ---------------------------------------------------------------------------
// Kernel 6: logp = logit - max - lse (in place); grid = (V/256, rows)
// ---------------------------------------------------------------------------
__global__ void norm_kernel(float* __restrict__ logits,
                            const float* __restrict__ rmax,
                            const float* __restrict__ rlse) {
    const int r = blockIdx.y;
    const int c = blockIdx.x * blockDim.x + threadIdx.x;
    logits[(size_t)r * VV + c] -= (rmax[r] + rlse[r]);
}

// ---------------------------------------------------------------------------
extern "C" void kernel_launch(void* const* d_in, const int* in_sizes, int n_in,
                              void* d_out, int out_size, void* d_ws, size_t ws_size,
                              hipStream_t stream) {
    const int*   tokens  = (const int*)  d_in[0];  // target_seqs [B,S]
    const float* thought = (const float*)d_in[1];  // [1,B,H]
    const float* hidden  = (const float*)d_in[2];  // [L,B,H]
    const float* emb     = (const float*)d_in[3];  // [V,H]
    const float* W_ih    = (const float*)d_in[4];  // [3H,2H]
    const float* W_hh    = (const float*)d_in[5];  // [3H,H]
    const float* b_ih    = (const float*)d_in[6];  // [3H]
    const float* b_hh    = (const float*)d_in[7];  // [3H]
    const float* W_out   = (const float*)d_in[8];  // [V,H]
    const float* b_out   = (const float*)d_in[9];  // [V]

    char* ws = (char*)d_ws;
    __bf16* X    = (__bf16*)(ws + 0);           // [1024,1024] bf16  = 2 MB
    float*  gi   = (float*) (ws + 2097152);     // [1024,1536] f32   = 6 MB
    __bf16* ys   = (__bf16*)(ws + 8388608);     // [1024, 512] bf16  = 1 MB
    float*  rmax = (float*) (ws + 9437184);     // [1024]
    float*  rlse = (float*) (ws + 9441280);     // [1024]

    float* logits = (float*)d_out;                       // [B,S,V]
    float* hlast  = logits + (size_t)BB * SS * VV;       // [1,B,H]

    // 1. gather + concat -> bf16 X  (rows ordered s*B+b)
    prep_x_kernel<<<MROWS, 256, 0, stream>>>(tokens, thought, emb, X);
    // 2. gi = X * W_ih^T + b_ih   (M=1024,N=1536,K=1024); LDS A-tile = 64 KB
    gemm_bf16_kernel<H2, 1><<<dim3(MROWS / 32, H3 / 64), 256, 32 * H2 * 2, stream>>>(
        X, W_ih, b_ih, gi, H3);
    // 3. GRU recurrence (single workgroup, LDS-resident hidden state)
    gru_kernel<<<1, 1024, BB * H3 * sizeof(float), stream>>>(
        gi, W_hh, b_hh, hidden, ys, hlast);
    // 4. logits = ys * W_out^T + b_out  (M=1024,N=32000,K=512); NT=4, LDS A = 32 KB
    gemm_bf16_kernel<HH, 4><<<dim3(MROWS / 32, VV / 256), 256, 32 * HH * 2, stream>>>(
        ys, W_out, b_out, logits, VV);
    // 5. per-row max / log-sum-exp
    row_lse_kernel<<<MROWS, 256, 0, stream>>>(logits, rmax, rlse);
    // 6. normalize in place -> log_softmax
    norm_kernel<<<dim3(VV / 256, MROWS), 256, 0, stream>>>(logits, rmax, rlse);
}
